// EnhancedTemporalGNN_76836964926248
// MI455X (gfx1250) — matrix operand order repaired and checked by reading, not compiled, not run on hardware
//
#include <hip/hip_runtime.h>
#include <hip/hip_bf16.h>

typedef __attribute__((ext_vector_type(16))) __bf16 v16bf;
typedef __attribute__((ext_vector_type(8)))  float  v8f;
typedef int v4i32 __attribute__((vector_size(16)));

#define DIM 128
#define HEADS 4
#define HD 32

#if defined(__AMDGCN__) && __has_builtin(__builtin_amdgcn_global_load_async_to_lds_b128)
#define HAVE_ASYNC_LDS 1
#else
#define HAVE_ASYNC_LDS 0
#endif

// ---------------- helpers ----------------
__device__ __forceinline__ unsigned fenc(float x) {
    unsigned b = __float_as_uint(x);
    return (b & 0x80000000u) ? ~b : (b | 0x80000000u);
}
__device__ __forceinline__ float fdec(unsigned e) {
    unsigned b = (e & 0x80000000u) ? (e ^ 0x80000000u) : ~e;
    return __uint_as_float(b);
}
__device__ __forceinline__ float wave_sum(float p) {
#pragma unroll
    for (int off = 16; off > 0; off >>= 1) p += __shfl_xor(p, off, 32);
    return p;
}
__device__ __forceinline__ void wait_async_lds() {
#if HAVE_ASYNC_LDS
#if __has_builtin(__builtin_amdgcn_s_wait_asynccnt)
    __builtin_amdgcn_s_wait_asynccnt(0);
#else
    asm volatile("s_wait_asynccnt 0" ::: "memory");
#endif
#endif
}

// ---------------- fused multi-output GEMM: out_g = (A0[+A1]) @ W_g + b_g ----------------
// A: M x 128 f32, W: 128 x 128 f32 (row-major, [k*128+n]), out: M x 128 f32.
// Block = 256 threads = 8 waves; block covers 128 rows; each wave a 16-row strip.
// bf16 WMMA 16x16x32 / f32 accumulate; W staged per-block in LDS (async-to-LDS when available).
__launch_bounds__(256)
__global__ void gemm128_multi(const float* __restrict__ A0, const float* __restrict__ A1,
                              int M,
                              const float* __restrict__ W0, const float* __restrict__ B0, float* __restrict__ O0,
                              const float* __restrict__ W1, const float* __restrict__ B1, float* __restrict__ O1,
                              const float* __restrict__ W2, const float* __restrict__ B2, float* __restrict__ O2,
                              const float* __restrict__ W3, const float* __restrict__ B3, float* __restrict__ O3,
                              int nW) {
    __shared__ float Wlf[DIM * DIM];   // 64 KB (f32 weights; bf16-convert at fragment build)
    const int tid  = threadIdx.x;
    const int wave = tid >> 5;
    const int lane = tid & 31;
    const int half = lane >> 4;        // 0 | 1
    const int mrow = lane & 15;        // M index for A rows; N index for B and C columns
    const long rowBase = (long)blockIdx.x * 128 + wave * 16;
    const long arow = rowBase + mrow;
    const long arowc = (arow < M) ? arow : 0;   // clamp OOB rows (stores are guarded)
    const bool fullWave = (rowBase + 16 <= M);  // wave-uniform store fast-path

    // ---- load A fragments for all 4 K-steps (ISA 16-bit A layout) ----
    v16bf afrag[4];
    {
        const float* ap0 = A0 + arowc * DIM;
        const float* ap1 = A1 ? (A1 + arowc * DIM) : nullptr;
#pragma unroll
        for (int ks = 0; ks < 4; ++ks) {
            const int kb = ks * 32 + half * 8;
#pragma unroll
            for (int j = 0; j < 8; ++j) {
                float f0 = ap0[kb + j];
                float f1 = ap0[kb + 16 + j];
                if (ap1) { f0 += ap1[kb + j]; f1 += ap1[kb + 16 + j]; }
                afrag[ks][j]     = (__bf16)f0;
                afrag[ks][8 + j] = (__bf16)f1;
            }
        }
    }

    // stage one 128x128 f32 weight matrix into LDS
    auto stage = [&](const float* __restrict__ W) {
        __syncthreads();               // protect prior readers of Wlf
#if HAVE_ASYNC_LDS
#pragma unroll
        for (int it = 0; it < 16; ++it) {
            const int i = it * 256 + tid;   // b128 chunk index, 4096 total
            __builtin_amdgcn_global_load_async_to_lds_b128(
                (v4i32 __attribute__((address_space(1)))*)((const char*)W + (size_t)i * 16),
                (v4i32 __attribute__((address_space(3)))*)((char*)Wlf + (size_t)i * 16),
                0, 0);
        }
        wait_async_lds();
#else
        const float4* Wv = (const float4*)W;
        float4* Lv = (float4*)Wlf;
        for (int i = tid; i < DIM * DIM / 4; i += 256) Lv[i] = Wv[i];
#endif
        __syncthreads();
    };

    // compute one output given staged W
    auto compute = [&](const float* __restrict__ Bv, float* __restrict__ Op) {
#pragma unroll
        for (int ct = 0; ct < 8; ++ct) {
            const int n = ct * 16 + mrow;          // output column
            const float bvv = Bv[n];
            v8f acc = {bvv, bvv, bvv, bvv, bvv, bvv, bvv, bvv};
#pragma unroll
            for (int ks = 0; ks < 4; ++ks) {
                // B fragment (ISA 16-bit B layout): lane holds col n, K = half*8 + j (j<8), 16 + half*8 + (j-8) (j>=8)
                v16bf bfrag;
                const int kb = ks * 32 + half * 8;
#pragma unroll
                for (int j = 0; j < 16; ++j)
                    bfrag[j] = (__bf16)Wlf[(kb + j + (j >= 8 ? 8 : 0)) * DIM + n];
                acc = __builtin_amdgcn_wmma_f32_16x16x32_bf16(
                    false, afrag[ks], false, bfrag, (short)0, acc, false, false);
            }
            float* po = Op + (rowBase + half * 8) * DIM + n;
            if (fullWave) {
#pragma unroll
                for (int r = 0; r < 8; ++r) po[r * DIM] = acc[r];
            } else {
#pragma unroll
                for (int r = 0; r < 8; ++r)
                    if (rowBase + half * 8 + r < M) po[r * DIM] = acc[r];
            }
        }
    };

    stage(W0); compute(B0, O0);
    if (nW > 1) { stage(W1); compute(B1, O1); }
    if (nW > 2) { stage(W2); compute(B2, O2); }
    if (nW > 3) { stage(W3); compute(B3, O3); }
}

// ---------------- init per-head max/sum + zero aggregator ----------------
__global__ void init_heads(unsigned* __restrict__ mxu, float* __restrict__ sm, int n) {
    int i = blockIdx.x * blockDim.x + threadIdx.x;
    if (i < n) { mxu[i] = 0x007FFFFFu; /* fenc(-inf) */ sm[i] = 0.0f; }
}
__global__ void zero_f(float* __restrict__ p, int n) {
    int i = blockIdx.x * blockDim.x + threadIdx.x;
    if (i < n) p[i] = 0.0f;
}

// ---------------- edge pass 1: logits + segment-max (atomic, order-preserving uint) ----------------
__global__ void edge_logits(const int* __restrict__ ei, const float* __restrict__ Q,
                            const float* __restrict__ K, const float* __restrict__ EW,
                            float* __restrict__ logits, unsigned* __restrict__ mxu, int E_) {
    int idx = blockIdx.x * blockDim.x + threadIdx.x;
    if (idx >= E_ * HEADS) return;
    const int e = idx >> 2, h = idx & 3;
    const int src = ei[e], dst = ei[E_ + e];
    const float4* q  = (const float4*)(Q  + (long)dst * DIM + h * HD);
    const float4* k  = (const float4*)(K  + (long)src * DIM + h * HD);
    const float4* ew = (const float4*)(EW + (long)e   * DIM + h * HD);
    float s = 0.0f;
#pragma unroll
    for (int i = 0; i < HD / 4; ++i) {
        float4 qv = q[i], kv = k[i], ev = ew[i];
        s += qv.x * (kv.x + ev.x) + qv.y * (kv.y + ev.y) +
             qv.z * (kv.z + ev.z) + qv.w * (kv.w + ev.w);
    }
    s *= 0.17677669529663687f;   // 32^-0.5
    logits[idx] = s;
    atomicMax(&mxu[(long)dst * HEADS + h], fenc(s));
}

// ---------------- edge pass 2: exp(logit - max) + segment-sum ----------------
__global__ void edge_expsum(const int* __restrict__ ei, float* __restrict__ logits,
                            const unsigned* __restrict__ mxu, float* __restrict__ sm, int E_) {
    int idx = blockIdx.x * blockDim.x + threadIdx.x;
    if (idx >= E_ * HEADS) return;
    const int e = idx >> 2, h = idx & 3;
    const int dst = ei[E_ + e];
    const float m  = fdec(mxu[(long)dst * HEADS + h]);
    const float ex = expf(logits[idx] - m);
    logits[idx] = ex;            // overwrite in place with exp
    atomicAdd(&sm[(long)dst * HEADS + h], ex);
}

// ---------------- edge pass 3: alpha-weighted scatter of (V[src]+EW) ----------------
__global__ void edge_scatter(const int* __restrict__ ei, const float* __restrict__ exl,
                             const float* __restrict__ sm, const float* __restrict__ V,
                             const float* __restrict__ EW, float* __restrict__ AGG, int E_) {
    int idx = blockIdx.x * blockDim.x + threadIdx.x;
    if (idx >= E_ * (DIM / 4)) return;
    const int e = idx >> 5;
    const int part = idx & 31;
    const int d0 = part * 4;
    const int h = d0 >> 5;       // which head this dim-group belongs to
    const int src = ei[e], dst = ei[E_ + e];
    const float alpha = exl[(long)e * HEADS + h] / (sm[(long)dst * HEADS + h] + 1e-16f);
    const float4 v  = *(const float4*)(V  + (long)src * DIM + d0);
    const float4 ew = *(const float4*)(EW + (long)e   * DIM + d0);
    float* ap = AGG + (long)dst * DIM + d0;
    atomicAdd(&ap[0], (v.x + ew.x) * alpha);
    atomicAdd(&ap[1], (v.y + ew.y) * alpha);
    atomicAdd(&ap[2], (v.z + ew.z) * alpha);
    atomicAdd(&ap[3], (v.w + ew.w) * alpha);
}

// ---------------- per-node: beta-gate combine + exact GELU + LayerNorm ----------------
__launch_bounds__(256)
__global__ void node_post(const float* __restrict__ AGG, const float* __restrict__ S,
                          const float* __restrict__ Wb, const float* __restrict__ bb,
                          const float* __restrict__ ln_g, const float* __restrict__ ln_b,
                          float* __restrict__ H, int Nn) {
    const int node = blockIdx.x * 8 + (threadIdx.x >> 5);
    const int lane = threadIdx.x & 31;
    if (node >= Nn) return;
    const int d = lane * 4;
    float4 o  = *(const float4*)(AGG + (long)node * DIM + d);
    float4 sk = *(const float4*)(S   + (long)node * DIM + d);

    // beta = sigmoid([out, skip, out-skip] @ Wb + bb)
    float p = o.x * Wb[d] + o.y * Wb[d + 1] + o.z * Wb[d + 2] + o.w * Wb[d + 3];
    p += sk.x * Wb[DIM + d] + sk.y * Wb[DIM + d + 1] + sk.z * Wb[DIM + d + 2] + sk.w * Wb[DIM + d + 3];
    p += (o.x - sk.x) * Wb[2 * DIM + d]     + (o.y - sk.y) * Wb[2 * DIM + d + 1] +
         (o.z - sk.z) * Wb[2 * DIM + d + 2] + (o.w - sk.w) * Wb[2 * DIM + d + 3];
    p = wave_sum(p);
    const float beta = 1.0f / (1.0f + expf(-(p + bb[0])));

    float h0 = beta * sk.x + (1.0f - beta) * o.x;
    float h1 = beta * sk.y + (1.0f - beta) * o.y;
    float h2 = beta * sk.z + (1.0f - beta) * o.z;
    float h3 = beta * sk.w + (1.0f - beta) * o.w;

    // exact GELU: 0.5*x*(1+erf(x/sqrt(2)))
    const float is2 = 0.7071067811865476f;
    h0 = 0.5f * h0 * (1.0f + erff(h0 * is2));
    h1 = 0.5f * h1 * (1.0f + erff(h1 * is2));
    h2 = 0.5f * h2 * (1.0f + erff(h2 * is2));
    h3 = 0.5f * h3 * (1.0f + erff(h3 * is2));

    // LayerNorm over 128 dims
    const float mean = wave_sum(h0 + h1 + h2 + h3) * (1.0f / 128.0f);
    const float d0f = h0 - mean, d1f = h1 - mean, d2f = h2 - mean, d3f = h3 - mean;
    const float var = wave_sum(d0f * d0f + d1f * d1f + d2f * d2f + d3f * d3f) * (1.0f / 128.0f);
    const float inv = rsqrtf(var + 1e-5f);

    float* hp = H + (long)node * DIM + d;
    hp[0] = d0f * inv * ln_g[d]     + ln_b[d];
    hp[1] = d1f * inv * ln_g[d + 1] + ln_b[d + 1];
    hp[2] = d2f * inv * ln_g[d + 2] + ln_b[d + 2];
    hp[3] = d3f * inv * ln_g[d + 3] + ln_b[d + 3];
}

// ---------------- final: g = sigmoid(x@Wg+bg); y = g*x + (1-g)*OUT ----------------
__launch_bounds__(256)
__global__ void final_gate(const float* __restrict__ x, const float* __restrict__ OUT,
                           const float* __restrict__ Wg, const float* __restrict__ bg,
                           float* __restrict__ y, int Nn) {
    const int node = blockIdx.x * 8 + (threadIdx.x >> 5);
    const int lane = threadIdx.x & 31;
    if (node >= Nn) return;
    const int d = lane * 4;
    float4 xv = *(const float4*)(x   + (long)node * DIM + d);
    float4 ov = *(const float4*)(OUT + (long)node * DIM + d);
    float p = xv.x * Wg[d] + xv.y * Wg[d + 1] + xv.z * Wg[d + 2] + xv.w * Wg[d + 3];
    p = wave_sum(p);
    const float g = 1.0f / (1.0f + expf(-(p + bg[0])));
    float* yp = y + (long)node * DIM + d;
    yp[0] = g * xv.x + (1.0f - g) * ov.x;
    yp[1] = g * xv.y + (1.0f - g) * ov.y;
    yp[2] = g * xv.z + (1.0f - g) * ov.z;
    yp[3] = g * xv.w + (1.0f - g) * ov.w;
}

extern "C" void kernel_launch(void* const* d_in, const int* in_sizes, int n_in,
                              void* d_out, int out_size, void* d_ws, size_t ws_size,
                              hipStream_t stream) {
    const float* x  = (const float*)d_in[0];
    const int*   ei = (const int*)d_in[1];
    const float* ea = (const float*)d_in[2];
    const int Nn = in_sizes[0] / DIM;
    const int Ee = in_sizes[1] / 2;

    // params: top-level insertion order {x, edge_index, edge_attr, params};
    // nested dicts flattened with sorted keys (JAX pytree order):
    // conv*: beta(W,b), e(W,b), k(W,b), q(W,b), skip(W,b), v(W,b)
    const int C1 = 3, C2 = 15;
    auto F = [&](int i) { return (const float*)d_in[i]; };

    // workspace carve-up (floats)
    float* ws = (float*)d_ws;
    size_t o = 0;
    float* EW  = ws + o; o += (size_t)Ee * DIM;
    float* Qb  = ws + o; o += (size_t)Nn * DIM;
    float* Kb  = ws + o; o += (size_t)Nn * DIM;
    float* Vb  = ws + o; o += (size_t)Nn * DIM;
    float* Sb  = ws + o; o += (size_t)Nn * DIM;
    float* AGG = ws + o; o += (size_t)Nn * DIM;
    float* Hb  = ws + o; o += (size_t)Nn * DIM;
    float* H2b = ws + o; o += (size_t)Nn * DIM;
    float* exl = ws + o; o += (size_t)Ee * HEADS;
    unsigned* mxu = (unsigned*)(ws + o); o += (size_t)Nn * HEADS;
    float* smb = ws + o; o += (size_t)Nn * HEADS;
    (void)ws_size;

    const int gridN  = (Nn + 127) / 128;
    const int gridE  = (Ee + 127) / 128;
    const int gridEH = (Ee * HEADS + 255) / 256;
    const int gridES = (Ee * (DIM / 4) + 255) / 256;
    const int gridNP = (Nn + 7) / 8;

    auto run_conv = [&](const float* feat, int base, const float* lng, const float* lnb,
                        float* Hout) {
        // node linears: q,k,v,skip fused (feat read once)
        gemm128_multi<<<gridN, 256, 0, stream>>>(
            feat, nullptr, Nn,
            F(base + 6), F(base + 7), Qb,    // q
            F(base + 4), F(base + 5), Kb,    // k
            F(base + 10), F(base + 11), Vb,  // v
            F(base + 8), F(base + 9), Sb,    // skip
            4);
        // edge linear
        gemm128_multi<<<gridE, 256, 0, stream>>>(
            ea, nullptr, Ee,
            F(base + 2), F(base + 3), EW,    // e
            nullptr, nullptr, nullptr, nullptr, nullptr, nullptr,
            nullptr, nullptr, nullptr, 1);
        init_heads<<<(Nn * HEADS + 255) / 256, 256, 0, stream>>>(mxu, smb, Nn * HEADS);
        zero_f<<<((int)((size_t)Nn * DIM) + 255) / 256, 256, 0, stream>>>(AGG, Nn * DIM);
        edge_logits<<<gridEH, 256, 0, stream>>>(ei, Qb, Kb, EW, exl, mxu, Ee);
        edge_expsum<<<gridEH, 256, 0, stream>>>(ei, exl, mxu, smb, Ee);
        edge_scatter<<<gridES, 256, 0, stream>>>(ei, exl, smb, Vb, EW, AGG, Ee);
        node_post<<<gridNP, 256, 0, stream>>>(AGG, Sb, F(base + 0), F(base + 1),
                                              lng, lnb, Hout, Nn);
    };

    // conv1 on x -> Hb ; conv2 on Hb -> H2b
    run_conv(x, C1, F(30), F(29), Hb);   // norm1: g=30, b=29
    run_conv(Hb, C2, F(32), F(31), H2b); // norm2: g=32, b=31

    // out = (H + H2) @ Wout + bout   (A = Hb + H2b fused in GEMM A-load); reuse Qb
    gemm128_multi<<<gridN, 256, 0, stream>>>(
        Hb, H2b, Nn,
        F(33), F(34), Qb,
        nullptr, nullptr, nullptr, nullptr, nullptr, nullptr,
        nullptr, nullptr, nullptr, 1);

    // y = sigmoid(x@Wg+bg)*x + (1-g)*out
    final_gate<<<gridNP, 256, 0, stream>>>(x, Qb, F(27), F(28), (float*)d_out, Nn);
}
